// RestrictedAttention_55748675502798
// MI455X (gfx1250) — matrix operand order repaired and checked by reading, not compiled, not run on hardware
//
#include <hip/hip_runtime.h>
#include <hip/hip_bf16.h>

// ---------------------------------------------------------------------------
// Banded multi-head attention for MI455X (gfx1250, wave32, WMMA).
//   FLOPs dominated by four 512x512 projections (~17 GFLOP) -> bf16 WMMA.
//   Attention band is 33 keys/query -> fused kernel, 48-col window per
//   16-query block, scores never materialized in HBM.
//   GEMM: 32x64 output tile per wave (8 accumulators, 2x A-frag reuse of
//   each weight fragment); global_prefetch_b8 ahead of the K-loop.
// ---------------------------------------------------------------------------

#define S_LEN   4096
#define D_MOD   512
#define N_HEADS 8
#define DK      64
#define BATCH   2
#define M_ROWS  (BATCH * S_LEN)   // 8192

typedef __bf16 bf16;
typedef __attribute__((ext_vector_type(16))) __bf16 v16bf;
typedef __attribute__((ext_vector_type(8)))  __bf16 bf16x8;
typedef __attribute__((ext_vector_type(8)))  float  v8f;
typedef __attribute__((ext_vector_type(4)))  float  f32x4;

__device__ __forceinline__ v16bf join8(bf16x8 lo, bf16x8 hi) {
  return __builtin_shufflevector(lo, hi, 0,1,2,3,4,5,6,7,8,9,10,11,12,13,14,15);
}

__device__ __forceinline__ v8f wmma_bf16(v16bf a, v16bf b, v8f c) {
  // (neg_a, A, neg_b, B, c_mod, C, reuse_a, reuse_b)
  return __builtin_amdgcn_wmma_f32_16x16x32_bf16(false, a, false, b, (short)0, c,
                                                 false, false);
}

// ---------------------------------------------------------------------------
// f32 -> bf16 conversion (weights only; activations converted on the fly)
// ---------------------------------------------------------------------------
__global__ __launch_bounds__(256) void cvt_f32_bf16(const float* __restrict__ src,
                                                    bf16* __restrict__ dst, int n) {
  int i = blockIdx.x * blockDim.x + threadIdx.x;
  if (i < n) dst[i] = (bf16)src[i];
}

// ---------------------------------------------------------------------------
// A-fragment loader (ISA 7.12.2 layout): lane(l<16) row M=l,
// halves 0..7 = K[k0+grp*8 ..], halves 8..15 = K[k0+16+grp*8 ..].
// MODE 0/1: f32 source converted on the fly; MODE 2: bf16 source.
// ---------------------------------------------------------------------------
template <int MODE>
__device__ __forceinline__ v16bf load_afrag(const float* __restrict__ Af,
                                            const bf16* __restrict__ Ab,
                                            int row, int k0, int grp) {
  if (MODE == 2) {
    const bf16* arow = Ab + (size_t)row * D_MOD;
    return join8(*reinterpret_cast<const bf16x8*>(arow + k0 + grp * 8),
                 *reinterpret_cast<const bf16x8*>(arow + k0 + 16 + grp * 8));
  } else {
    const float* arow = Af + (size_t)row * D_MOD;
    f32x4 x0 = *reinterpret_cast<const f32x4*>(arow + k0 + grp * 8);
    f32x4 x1 = *reinterpret_cast<const f32x4*>(arow + k0 + grp * 8 + 4);
    f32x4 x2 = *reinterpret_cast<const f32x4*>(arow + k0 + 16 + grp * 8);
    f32x4 x3 = *reinterpret_cast<const f32x4*>(arow + k0 + 16 + grp * 8 + 4);
    v16bf a;
#pragma unroll
    for (int i = 0; i < 4; ++i) {
      a[i]      = (bf16)x0[i];
      a[4 + i]  = (bf16)x1[i];
      a[8 + i]  = (bf16)x2[i];
      a[12 + i] = (bf16)x3[i];
    }
    return a;
  }
}

// ---------------------------------------------------------------------------
// Generic projection GEMM:  C[M, 512] = A[M, 512] @ W^T + bias
//   W is torch layout [out, in] = [512, 512] row-major (already bf16).
//   MODE 0: A = f32 activations, out = bf16 [M,512], value scaled by `scale`
//   MODE 1: A = f32 activations, out = bf16 transposed-per-head [B,H,DK,S]
//   MODE 2: A = bf16 ctx,        out = f32  [M,512]  (final projection)
// One wave computes a 32x64 output tile (two 16-row halves share each
// weight fragment); K-loop in steps of 32.
// B-fragment layout: lane(l<16) col N=l, halves 0..7 = K[k0+grp*8 ..],
// halves 8..15 = K[k0+16+grp*8 ..]. C: lane col N=l%16, vgpr r ->
// row M = grp*8 + r.
// ---------------------------------------------------------------------------
template <int MODE>
__global__ __launch_bounds__(256) void gemm_proj(const void* __restrict__ Aptr,
                                                 const bf16* __restrict__ W,
                                                 const float* __restrict__ bias,
                                                 void* __restrict__ Out,
                                                 float scale) {
  const int lane = threadIdx.x & 31;
  const int grp  = lane >> 4;
  const int lr   = lane & 15;
  const int wid  = blockIdx.x * (blockDim.x >> 5) + (threadIdx.x >> 5);
  const int NT   = D_MOD / 64;                 // 8 column groups of 64
  const int m0   = (wid / NT) * 32;
  const int n0   = (wid % NT) * 64;
  if (m0 >= M_ROWS) return;

  const float* Af = (const float*)Aptr;
  const bf16*  Ab = (const bf16*)Aptr;

  v8f acc[8] = {v8f{}, v8f{}, v8f{}, v8f{}, v8f{}, v8f{}, v8f{}, v8f{}};

  const int am0 = m0 + lr;
  const int am1 = m0 + 16 + lr;
  for (int k0 = 0; k0 < D_MOD; k0 += 32) {
    // Prefetch next K-step of the activation rows (global_prefetch_b8).
    if (k0 + 32 < D_MOD) {
      if (MODE == 2) {
        __builtin_prefetch(Ab + (size_t)am0 * D_MOD + k0 + 32, 0, 1);
        __builtin_prefetch(Ab + (size_t)am1 * D_MOD + k0 + 32, 0, 1);
      } else {
        __builtin_prefetch(Af + (size_t)am0 * D_MOD + k0 + 32, 0, 1);
        __builtin_prefetch(Af + (size_t)am1 * D_MOD + k0 + 32, 0, 1);
      }
    }
    v16bf a0 = load_afrag<MODE>(Af, Ab, am0, k0, grp);
    v16bf a1 = load_afrag<MODE>(Af, Ab, am1, k0, grp);
#pragma unroll
    for (int t = 0; t < 4; ++t) {
      const int n = n0 + t * 16 + lr;
      const bf16* wrow = W + (size_t)n * D_MOD + k0 + grp * 8;
      v16bf bfrag = join8(*reinterpret_cast<const bf16x8*>(wrow),
                          *reinterpret_cast<const bf16x8*>(wrow + 16));
      acc[t]     = wmma_bf16(a0, bfrag, acc[t]);
      acc[4 + t] = wmma_bf16(a1, bfrag, acc[4 + t]);
    }
  }

  if (MODE == 0) {
    bf16* out = (bf16*)Out;
#pragma unroll
    for (int half = 0; half < 2; ++half) {
#pragma unroll
      for (int t = 0; t < 4; ++t) {
        const int n = n0 + t * 16 + lr;
        const float bv = bias[n];
#pragma unroll
        for (int r = 0; r < 8; ++r) {
          const int m = m0 + half * 16 + grp * 8 + r;
          out[(size_t)m * D_MOD + n] = (bf16)((acc[half * 4 + t][r] + bv) * scale);
        }
      }
    }
  } else if (MODE == 1) {
    // V^T layout: [B, H, DK, S]; 8 consecutive rows -> contiguous along S.
    bf16* out = (bf16*)Out;
    const int b = m0 >> 12;                      // 4096 rows per batch
#pragma unroll
    for (int half = 0; half < 2; ++half) {
      const int sbase = (m0 & (S_LEN - 1)) + half * 16 + grp * 8;
#pragma unroll
      for (int t = 0; t < 4; ++t) {
        const int n = n0 + t * 16 + lr;
        const int h = n >> 6, c = n & 63;
        const float bv = bias[n];
        bf16x8 vvec;
#pragma unroll
        for (int r = 0; r < 8; ++r) vvec[r] = (bf16)(acc[half * 4 + t][r] + bv);
        *reinterpret_cast<bf16x8*>(
            out + (((size_t)(b * N_HEADS + h) * DK + c) * S_LEN + sbase)) = vvec;
      }
    }
  } else {
    float* out = (float*)Out;
#pragma unroll
    for (int half = 0; half < 2; ++half) {
#pragma unroll
      for (int t = 0; t < 4; ++t) {
        const int n = n0 + t * 16 + lr;
        const float bv = bias[n];
#pragma unroll
        for (int r = 0; r < 8; ++r) {
          const int m = m0 + half * 16 + grp * 8 + r;
          out[(size_t)m * D_MOD + n] = acc[half * 4 + t][r] + bv;
        }
      }
    }
  }
}

// ---------------------------------------------------------------------------
// Fused banded attention. One wave per 16-query block per (b,h).
// Key window per block: kb = q0-16 .. q0+31 (48 cols, segment-aligned so
// fringe 16-key segments are entirely in- or out-of-range -> clamp + P=0).
// Q was pre-scaled by 1/sqrt(dk) at projection time.
// ---------------------------------------------------------------------------
__global__ __launch_bounds__(256) void attn_band(const bf16* __restrict__ Q,
                                                 const bf16* __restrict__ K,
                                                 const bf16* __restrict__ Vt,
                                                 bf16* __restrict__ Ctx) {
  __shared__ float sS[8][16][48];     // raw scores
  __shared__ bf16  sP[8][16][64];     // exp probs (unnormalized, zero-padded)
  __shared__ float sInv[8][16];       // 1/rowsum

  const int lane = threadIdx.x & 31;
  const int grp  = lane >> 4;
  const int lr   = lane & 15;
  const int w    = threadIdx.x >> 5;
  const int wid  = blockIdx.x * 8 + w;
  const int qblk = wid & 255;                 // 256 query blocks per (b,h)
  const int bh   = wid >> 8;                  // 0..15
  const int b    = bh >> 3, h = bh & 7;
  const int q0   = qblk * 16;
  const int kb   = q0 - 16;

  // ---- scores: 16x48 via 3 N-tiles x 2 K-steps of WMMA -------------------
  v8f sc[3] = {v8f{}, v8f{}, v8f{}};
#pragma unroll
  for (int k0 = 0; k0 < DK; k0 += 32) {
    const bf16* qrow = Q + ((size_t)(b * S_LEN + q0 + lr)) * D_MOD + h * DK;
    v16bf afrag = join8(*reinterpret_cast<const bf16x8*>(qrow + k0 + grp * 8),
                        *reinterpret_cast<const bf16x8*>(qrow + k0 + 16 + grp * 8));
#pragma unroll
    for (int t = 0; t < 3; ++t) {
      int key = kb + t * 16 + lr;
      key = min(max(key, 0), S_LEN - 1);      // OOB segments masked later
      const bf16* krow = K + ((size_t)(b * S_LEN + key)) * D_MOD + h * DK;
      v16bf bfrag = join8(
          *reinterpret_cast<const bf16x8*>(krow + k0 + grp * 8),
          *reinterpret_cast<const bf16x8*>(krow + k0 + 16 + grp * 8));
      sc[t] = wmma_bf16(afrag, bfrag, sc[t]);
    }
  }

#pragma unroll
  for (int t = 0; t < 3; ++t)
#pragma unroll
    for (int r = 0; r < 8; ++r)
      sS[w][grp * 8 + r][t * 16 + lr] = sc[t][r];
  __syncthreads();

  // ---- masked softmax, one row per lane (lanes 0..15) --------------------
  if (lane < 16) {
    const int row = lane;
    float mx = -3.0e38f;
    for (int j = 0; j < 48; ++j) {
      const int key = kb + j;
      const bool valid =
          (key >= 0) && (key < S_LEN) && (j >= row) && (j <= row + 32);
      float s = valid ? sS[w][row][j] : -1.0e30f;
      mx = fmaxf(mx, s);
    }
    float sum = 0.0f;
    for (int j = 0; j < 48; ++j) {
      const int key = kb + j;
      const bool valid =
          (key >= 0) && (key < S_LEN) && (j >= row) && (j <= row + 32);
      float e = valid ? __expf(sS[w][row][j] - mx) : 0.0f;
      sum += e;
      sP[w][row][j] = (bf16)e;
    }
    for (int j = 48; j < 64; ++j) sP[w][row][j] = (bf16)0.0f;
    sInv[w][row] = 1.0f / sum;
  }
  __syncthreads();

  // ---- ctx = P @ V : 4 dk-tiles x 2 K-steps (K padded 48->64, P=0 pad) ---
  v8f cacc[4] = {v8f{}, v8f{}, v8f{}, v8f{}};
#pragma unroll
  for (int k0 = 0; k0 < 64; k0 += 32) {
    const bf16* prow = &sP[w][lr][0];
    v16bf pfrag = join8(*reinterpret_cast<const bf16x8*>(prow + k0 + grp * 16),
                        *reinterpret_cast<const bf16x8*>(prow + k0 + grp * 16 + 8));
#pragma unroll
    for (int t = 0; t < 4; ++t) {
      const int c = t * 16 + lr;
      int sb = kb + k0 + grp * 16;
      sb = min(max(sb, 0), S_LEN - 16);       // OOB segments have P=0
      const bf16* vrow = Vt + (((size_t)(b * N_HEADS + h) * DK + c) * S_LEN + sb);
      v16bf bfrag = join8(*reinterpret_cast<const bf16x8*>(vrow),
                          *reinterpret_cast<const bf16x8*>(vrow + 8));
      cacc[t] = wmma_bf16(pfrag, bfrag, cacc[t]);
    }
  }

#pragma unroll
  for (int t = 0; t < 4; ++t) {
    const int c = t * 16 + lr;
#pragma unroll
    for (int r = 0; r < 8; ++r) {
      const int row = grp * 8 + r;
      const float inv = sInv[w][row];
      Ctx[((size_t)(b * S_LEN + q0 + row)) * D_MOD + h * DK + c] =
          (bf16)(cacc[t][r] * inv);
    }
  }
}

// ---------------------------------------------------------------------------
extern "C" void kernel_launch(void* const* d_in, const int* in_sizes, int n_in,
                              void* d_out, int out_size, void* d_ws,
                              size_t ws_size, hipStream_t stream) {
  const float* q  = (const float*)d_in[0];
  const float* k  = (const float*)d_in[1];
  const float* v  = (const float*)d_in[2];
  const float* wq = (const float*)d_in[3];
  const float* bq = (const float*)d_in[4];
  const float* wk = (const float*)d_in[5];
  const float* bk = (const float*)d_in[6];
  const float* wv = (const float*)d_in[7];
  const float* bv = (const float*)d_in[8];
  const float* wo = (const float*)d_in[9];
  const float* bo = (const float*)d_in[10];

  char* ws = (char*)d_ws;
  const size_t WBYTES = (size_t)D_MOD * D_MOD * sizeof(bf16);      // 512 KB
  const size_t TBYTES = (size_t)M_ROWS * D_MOD * sizeof(bf16);     // 8 MB
  bf16* wqb = (bf16*)(ws + 0 * WBYTES);
  bf16* wkb = (bf16*)(ws + 1 * WBYTES);
  bf16* wvb = (bf16*)(ws + 2 * WBYTES);
  bf16* wob = (bf16*)(ws + 3 * WBYTES);
  bf16* Qb  = (bf16*)(ws + 4 * WBYTES);
  bf16* Kb  = (bf16*)(ws + 4 * WBYTES + 1 * TBYTES);
  bf16* Vt  = (bf16*)(ws + 4 * WBYTES + 2 * TBYTES);
  bf16* Ctx = (bf16*)(ws + 4 * WBYTES + 3 * TBYTES);

  const int WN = D_MOD * D_MOD;                        // 262144
  const int cvtBlocks = (WN + 255) / 256;
  cvt_f32_bf16<<<cvtBlocks, 256, 0, stream>>>(wq, wqb, WN);
  cvt_f32_bf16<<<cvtBlocks, 256, 0, stream>>>(wk, wkb, WN);
  cvt_f32_bf16<<<cvtBlocks, 256, 0, stream>>>(wv, wvb, WN);
  cvt_f32_bf16<<<cvtBlocks, 256, 0, stream>>>(wo, wob, WN);

  // waves = (M/32)*(N/64) = 256*8 = 2048 ; 8 waves per 256-thread block
  const int gemmBlocks = (M_ROWS / 32) * (D_MOD / 64) / 8;  // 256
  const float qscale = 0.125f;  // 1/sqrt(dk)
  gemm_proj<0><<<gemmBlocks, 256, 0, stream>>>(q, wqb, bq, Qb, qscale);
  gemm_proj<0><<<gemmBlocks, 256, 0, stream>>>(k, wkb, bk, Kb, 1.0f);
  gemm_proj<1><<<gemmBlocks, 256, 0, stream>>>(v, wvb, bv, Vt, 1.0f);

  // waves = B*H*(S/16) = 4096 ; 8 per block
  attn_band<<<512, 256, 0, stream>>>(Qb, Kb, Vt, Ctx);

  gemm_proj<2><<<gemmBlocks, 256, 0, stream>>>(Ctx, wob, bo, d_out, 1.0f);
}
